// MolecularGraphNeuralNetwork_record_52209622450807
// MI455X (gfx1250) — matrix operand order repaired and checked by reading, not compiled, not run on hardware
//
#include <hip/hip_runtime.h>
#include <hip/hip_bf16.h>
#include <stdint.h>

typedef __attribute__((ext_vector_type(16))) _Float16 v16h;
typedef __attribute__((ext_vector_type(8)))  _Float16 v8h;
typedef __attribute__((ext_vector_type(8)))  float    v8f;

#define DIM      128
#define NMED     512
#define NNODES   32768
#define NEDGES   524288
#define NDIAG    2000
#define NPRO     1500
#define KP_DIAG  2016       // 2000 padded to /32
#define KP_PRO   1504       // 1500 padded to /32
#define FAGCN_EPS 0.3f

enum Epi { EPI_NONE = 0, EPI_BIAS = 1, EPI_BIAS_TANH = 2, EPI_SEL = 3 };
// OFLAGS bits: 1 = also write f16 row-major copy, 2 = also write f16
// transposed copy, 4 = skip the f32 C store.

// ---------------------------------------------------------------------------
// f32 -> zero-padded f16 operand conversion (row-major and transposing).
// ---------------------------------------------------------------------------
__global__ void cvt16_row_kernel(const float* __restrict__ in,
                                 _Float16* __restrict__ out,
                                 int M, int K, int KP) {
  int i = blockIdx.x * blockDim.x + threadIdx.x;
  if (i >= M * KP) return;
  int m = i / KP, k = i - m * KP;
  out[i] = (k < K) ? (_Float16)in[(size_t)m * K + k] : (_Float16)0.0f;
}

__global__ void cvt16_trans_kernel(const float* __restrict__ in,  // [K,N] f32
                                   _Float16* __restrict__ out,    // [N,KP] f16
                                   int K, int N, int KP) {
  int i = blockIdx.x * blockDim.x + threadIdx.x;
  if (i >= N * KP) return;
  int n = i / KP, k = i - n * KP;
  out[i] = (k < K) ? (_Float16)in[(size_t)k * N + n] : (_Float16)0.0f;
}

// ---------------------------------------------------------------------------
// WMMA GEMM, all-f16 operands / f32 accumulate.
//   A: [M, KP] f16 row-major.   B: [N, KP] f16 row-major (i.e. B^T layout):
//   C[m,n] = sum_k A[m,k] * B[n,k].  KP % 32 == 0, zero-padded -> the inner
//   loop is fully branchless: 4x global_load_b128 + 1x v_wmma per K-step.
// Fragment layouts used by this code (wave32):
//   A 16x32: row = lane&15; K run  [kb+16*hl, +16)
//   B 32x16: col = lane&15; K run  [kb+16*hl, +16)
//   C 16x16: col = lane&15; vgpr r -> row = r + 8*hl
// One wave per 16x16 tile, 4 waves / block.
// ---------------------------------------------------------------------------
template <int EPI, int OFLAGS>
__global__ __launch_bounds__(128)
void wmma_gemm16_kernel(const _Float16* __restrict__ A,
                        const _Float16* __restrict__ B,
                        const float* __restrict__ bias,
                        const float* __restrict__ extra1,   // EPI_SEL base
                        const float* __restrict__ extra2,   // EPI_SEL mult
                        float* __restrict__ C, int ldc,
                        _Float16* __restrict__ crow, int ldrow,
                        _Float16* __restrict__ ctrans, int ldtr,
                        int M, int N, int KP) {
  const int lane = threadIdx.x & 31;
  const int wave = threadIdx.x >> 5;
  const int tile = blockIdx.x * 4 + wave;
  const int tnc  = N >> 4;
  const int tm   = tile / tnc;
  const int tn   = tile % tnc;
  if (tm * 16 >= M) return;                 // wave-uniform

  const int hl  = lane >> 4;
  const int l16 = lane & 15;
  const _Float16* __restrict__ ap = A + (size_t)(tm * 16 + l16) * KP;
  const _Float16* __restrict__ bp = B + (size_t)(tn * 16 + l16) * KP;

  v8f acc = {};
  for (int kb = 0; kb < KP; kb += 32) {
    v16h af = *(const v16h*)(ap + kb + 16 * hl);     // 2x global_load_b128
    v16h bf = *(const v16h*)(bp + kb + 16 * hl);     // 2x global_load_b128
    acc = __builtin_amdgcn_wmma_f32_16x16x32_f16(false, af, false, bf,
                                                 (short)0, acc, false, false);
  }

  const int bcol = tn * 16 + l16;
  float bb = 0.0f;
  if constexpr (EPI != EPI_NONE) bb = bias[bcol];
#pragma unroll
  for (int r = 0; r < 8; ++r) {
    const int m = tm * 16 + r + 8 * hl;
    float v = acc[r] + bb;
    if constexpr (EPI == EPI_BIAS_TANH) v = tanhf(v);
    if constexpr (EPI == EPI_SEL) {
      const float s = tanhf(v);
      const size_t idx = (size_t)m * ldc + bcol;
      v = extra1[idx] + s * extra2[idx];
    }
    if constexpr (!(OFLAGS & 4)) C[(size_t)m * ldc + bcol] = v;
    if constexpr (OFLAGS & 1) crow[(size_t)m * ldrow + bcol] = (_Float16)v;
    if constexpr (OFLAGS & 2) ctrans[(size_t)bcol * ldtr + m] = (_Float16)v;
  }
}

// ------------------------- graph-stage kernels -----------------------------

__global__ void degree_kernel(const int* __restrict__ dst,
                              float* __restrict__ deg, int n) {
  int e = blockIdx.x * blockDim.x + threadIdx.x;
  if (e < n) atomicAdd(&deg[dst[e]], 1.0f);
}

__global__ void norm_kernel(float* __restrict__ deg, int n) {
  int i = blockIdx.x * blockDim.x + threadIdx.x;
  if (i < n) deg[i] = rsqrtf(fmaxf(deg[i], 1.0f));
}

__global__ void gather_kernel(const float* __restrict__ table,
                              const int* __restrict__ fp,
                              float* __restrict__ h, int n) {
  int node = blockIdx.x * 8 + (threadIdx.x >> 5);
  int lane = threadIdx.x & 31;
  if (node >= n) return;
  const float4* s = (const float4*)(table + (size_t)fp[node] * DIM);
  ((float4*)(h + (size_t)node * DIM))[lane] = s[lane];
}

// One wave per edge; lane handles 4 of 128 channels (coalesced float4 rows).
// g = tanh(hd.gw[0:128] + hs.gw[128:256] + gb) via wave32 butterfly; then
// z[dst] += g*norm[d]*norm[s]*hs with f32 global atomics.
__global__ __launch_bounds__(256)
void edge_kernel(const float* __restrict__ h, const int* __restrict__ src,
                 const int* __restrict__ dst, const float* __restrict__ norm,
                 const float* __restrict__ gw, const float* __restrict__ gb,
                 float* __restrict__ z, int nedges) {
  int e = blockIdx.x * 8 + (threadIdx.x >> 5);
  if (e >= nedges) return;
  int lane = threadIdx.x & 31;
  int s = src[e], d = dst[e];
  float4 hsv = ((const float4*)(h + (size_t)s * DIM))[lane];
  float4 hdv = ((const float4*)(h + (size_t)d * DIM))[lane];
  float4 wd  = ((const float4*)gw)[lane];          // applies to h_dst
  float4 wsc = ((const float4*)(gw + DIM))[lane];  // applies to h_src
  float p = hdv.x * wd.x  + hdv.y * wd.y  + hdv.z * wd.z  + hdv.w * wd.w +
            hsv.x * wsc.x + hsv.y * wsc.y + hsv.z * wsc.z + hsv.w * wsc.w;
#pragma unroll
  for (int off = 16; off > 0; off >>= 1) p += __shfl_xor(p, off, 32);
  float g    = tanhf(p + gb[0]);
  float coef = g * norm[d] * norm[s];
  float* zr = z + (size_t)d * DIM + lane * 4;
  atomicAdd(zr + 0, coef * hsv.x);
  atomicAdd(zr + 1, coef * hsv.y);
  atomicAdd(zr + 2, coef * hsv.z);
  atomicAdd(zr + 3, coef * hsv.w);
}

__global__ void update_kernel(float* __restrict__ h, const float* __restrict__ z,
                              const float* __restrict__ attn,
                              const int* __restrict__ mol_size_p, int total) {
  int i = blockIdx.x * blockDim.x + threadIdx.x;
  if (i >= total) return;
  int node = i >> 7, c = i & (DIM - 1);
  int med  = node / mol_size_p[0];
  float v = FAGCN_EPS * h[i] + attn[med * DIM + c] * z[i];
  h[i] = fmaxf(v, 0.0f);
}

// Pool 64 atoms per molecule; emit directly as transposed f16 B operand
// [DIM, NMED] for the final WMMA projection.
__global__ void pool_kernel(const float* __restrict__ h,
                            _Float16* __restrict__ molT,
                            const int* __restrict__ mol_size_p, int nmed) {
  int med = blockIdx.x, c = threadIdx.x;
  int ms = mol_size_p[0];
  const float* base = h + (size_t)med * ms * DIM + c;
  float s = 0.0f;
  for (int a = 0; a < ms; ++a) s += base[(size_t)a * DIM];
  molT[(size_t)c * nmed + med] = (_Float16)s;
}

// ---------------------------------------------------------------------------

static inline char* aln256(char* p) {
  return (char*)(((uintptr_t)p + 255) & ~(uintptr_t)255);
}

extern "C" void kernel_launch(void* const* d_in, const int* in_sizes, int n_in,
                              void* d_out, int out_size, void* d_ws, size_t ws_size,
                              hipStream_t stream) {
  const int*   fp        = (const int*)  d_in[0];
  const int*   src       = (const int*)  d_in[1];
  const int*   dst       = (const int*)  d_in[2];
  const float* embed     = (const float*)d_in[3];
  const float* gate_w0   = (const float*)d_in[4];
  const float* gate_b0   = (const float*)d_in[5];
  const float* gate_w1   = (const float*)d_in[6];
  const float* gate_b1   = (const float*)d_in[7];
  const float* ctx_w     = (const float*)d_in[8];
  const float* ctx_b     = (const float*)d_in[9];
  const float* diag_tab  = (const float*)d_in[10];
  const float* pro_tab   = (const float*)d_in[11];
  const float* med2diag  = (const float*)d_in[12];
  const float* med2pro   = (const float*)d_in[13];
  const float* ehradj    = (const float*)d_in[14];
  const float* viewcat_w = (const float*)d_in[15];
  const float* viewcat_b = (const float*)d_in[16];
  const float* sel_w     = (const float*)d_in[17];
  const float* sel_b     = (const float*)d_in[18];
  const float* avg_proj  = (const float*)d_in[19];
  const int*   mol_size  = (const int*)  d_in[20];

  // ---- workspace partition ----
  char* ws = (char*)d_ws;
#define WSALLOC(ptr, type, nelem) \
  type* ptr = (type*)ws; ws = aln256(ws + (size_t)(nelem) * sizeof(type))

  WSALLOC(A_md,   _Float16, (size_t)NMED * KP_DIAG);  // med2diag   f16 padded
  WSALLOC(B_dt,   _Float16, (size_t)DIM * KP_DIAG);   // diag_tab^T f16 padded
  WSALLOC(A_mp,   _Float16, (size_t)NMED * KP_PRO);   // med2pro    f16 padded
  WSALLOC(B_pt,   _Float16, (size_t)DIM * KP_PRO);    // pro_tab^T  f16 padded
  WSALLOC(A_ehr,  _Float16, (size_t)NMED * NMED);     // ehradj     f16
  WSALLOC(B_vc,   _Float16, (size_t)DIM * 2 * DIM);   // viewcat_w  f16
  WSALLOC(B_sel,  _Float16, (size_t)DIM * DIM);       // sel_w      f16
  WSALLOC(B_ctx,  _Float16, (size_t)DIM * DIM);       // ctx_w      f16
  WSALLOC(A_avg,  _Float16, (size_t)NMED * NMED);     // avg_proj   f16
  WSALLOC(Acat,   _Float16, (size_t)NMED * 2 * DIM);  // cat(diag,pro) f16
  WSALLOC(A_mr0,  _Float16, (size_t)NMED * DIM);      // med_rec0 row f16
  WSALLOC(B_mr0T, _Float16, (size_t)DIM * NMED);      // med_rec0^T  f16
  WSALLOC(A_mr,   _Float16, (size_t)NMED * DIM);      // med_rec row f16
  WSALLOC(molT,   _Float16, (size_t)DIM * NMED);      // mol^T       f16
  WSALLOC(med_rec0, float,  (size_t)NMED * DIM);
  WSALLOC(aug,      float,  (size_t)NMED * DIM);
  WSALLOC(attn,     float,  (size_t)NMED * DIM);
  WSALLOC(normv,    float,  (size_t)NNODES);
  WSALLOC(h,        float,  (size_t)NNODES * DIM);
  WSALLOC(z,        float,  (size_t)NNODES * DIM);
#undef WSALLOC

  // ---- operand conversion (f32 -> padded f16) ----
#define CVT_R(in, out, M, K, KP) \
  cvt16_row_kernel<<<((M) * (KP) + 255) / 256, 256, 0, stream>>>(in, out, M, K, KP)
#define CVT_T(in, out, K, N, KP) \
  cvt16_trans_kernel<<<((N) * (KP) + 255) / 256, 256, 0, stream>>>(in, out, K, N, KP)
  CVT_R(med2diag,  A_md,  NMED, NDIAG, KP_DIAG);
  CVT_T(diag_tab,  B_dt,  NDIAG, DIM,  KP_DIAG);
  CVT_R(med2pro,   A_mp,  NMED, NPRO,  KP_PRO);
  CVT_T(pro_tab,   B_pt,  NPRO,  DIM,  KP_PRO);
  CVT_R(ehradj,    A_ehr, NMED, NMED,  NMED);
  CVT_R(viewcat_w, B_vc,  DIM,  2 * DIM, 2 * DIM);
  CVT_R(sel_w,     B_sel, DIM,  DIM,   DIM);
  CVT_R(ctx_w,     B_ctx, DIM,  DIM,   DIM);
  CVT_R(avg_proj,  A_avg, NMED, NMED,  NMED);
#undef CVT_R
#undef CVT_T

  const dim3 gblk(128);
  const dim3 ggrid((NMED / 16) * (DIM / 16) / 4);   // 256 tiles / 4 waves

  // med_diag -> Acat[:, :128] (f16 only)
  wmma_gemm16_kernel<EPI_NONE, 5><<<ggrid, gblk, 0, stream>>>(
      A_md, B_dt, nullptr, nullptr, nullptr, nullptr, DIM,
      Acat, 2 * DIM, nullptr, 0, NMED, DIM, KP_DIAG);
  // med_pro -> Acat[:, 128:] (f16 only)
  wmma_gemm16_kernel<EPI_NONE, 5><<<ggrid, gblk, 0, stream>>>(
      A_mp, B_pt, nullptr, nullptr, nullptr, nullptr, DIM,
      Acat + DIM, 2 * DIM, nullptr, 0, NMED, DIM, KP_PRO);
  // med_rec0 = Acat @ viewcat_w.T + b  (f32 + f16 row + f16 trans copies)
  wmma_gemm16_kernel<EPI_BIAS, 3><<<ggrid, gblk, 0, stream>>>(
      Acat, B_vc, viewcat_b, nullptr, nullptr, med_rec0, DIM,
      A_mr0, DIM, B_mr0T, NMED, NMED, DIM, 2 * DIM);
  // aug = ehradj @ med_rec0
  wmma_gemm16_kernel<EPI_NONE, 0><<<ggrid, gblk, 0, stream>>>(
      A_ehr, B_mr0T, nullptr, nullptr, nullptr, aug, DIM,
      nullptr, 0, nullptr, 0, NMED, DIM, NMED);
  // med_rec = med_rec0 + tanh(med_rec0 @ sel_w.T + sel_b) * aug  (f16 row only)
  wmma_gemm16_kernel<EPI_SEL, 5><<<ggrid, gblk, 0, stream>>>(
      A_mr0, B_sel, sel_b, med_rec0, aug, nullptr, DIM,
      A_mr, DIM, nullptr, 0, NMED, DIM, DIM);
  // attn = tanh(med_rec @ ctx_w.T + ctx_b)
  wmma_gemm16_kernel<EPI_BIAS_TANH, 0><<<ggrid, gblk, 0, stream>>>(
      A_mr, B_ctx, ctx_b, nullptr, nullptr, attn, DIM,
      nullptr, 0, nullptr, 0, NMED, DIM, DIM);

  // degree -> norm
  hipMemsetAsync(normv, 0, (size_t)NNODES * 4, stream);
  degree_kernel<<<NEDGES / 256, 256, 0, stream>>>(dst, normv, NEDGES);
  norm_kernel<<<NNODES / 256, 256, 0, stream>>>(normv, NNODES);

  // h = embed_table[fingerprints]
  gather_kernel<<<NNODES / 8, 256, 0, stream>>>(embed, fp, h, NNODES);

  // two FAGCN layers
  const float* gws[2] = {gate_w0, gate_w1};
  const float* gbs[2] = {gate_b0, gate_b1};
  for (int layer = 0; layer < 2; ++layer) {
    hipMemsetAsync(z, 0, (size_t)NNODES * DIM * 4, stream);
    edge_kernel<<<NEDGES / 8, 256, 0, stream>>>(h, src, dst, normv,
                                                gws[layer], gbs[layer], z, NEDGES);
    update_kernel<<<(NNODES * DIM) / 256, 256, 0, stream>>>(h, z, attn, mol_size,
                                                            NNODES * DIM);
  }

  // pooling (emits transposed f16 B operand) + final projection
  pool_kernel<<<NMED, DIM, 0, stream>>>(h, molT, mol_size, NMED);
  wmma_gemm16_kernel<EPI_NONE, 0><<<ggrid, gblk, 0, stream>>>(
      A_avg, molT, nullptr, nullptr, nullptr, (float*)d_out, DIM,
      nullptr, 0, nullptr, 0, NMED, DIM, NMED);

  (void)in_sizes; (void)n_in; (void)out_size; (void)ws_size;
}
